// GCNShadowModel_20349555048515
// MI455X (gfx1250) — matrix operand order repaired and checked by reading, not compiled, not run on hardware
//
#include <hip/hip_runtime.h>
#include <hip/hip_bf16.h>

typedef __attribute__((ext_vector_type(16))) __bf16 v16bf;
typedef __attribute__((ext_vector_type(8)))  float  v8f;

// ---------------------------------------------------------------------------
// Degree / normalization
// ---------------------------------------------------------------------------
__global__ void deg_init_kernel(float* __restrict__ deg, int N) {
    int i = blockIdx.x * blockDim.x + threadIdx.x;
    if (i < N) deg[i] = 1.0f;  // self-loop contributes 1 to every node's degree
}

__global__ void deg_accum_kernel(const int* __restrict__ dst, float* __restrict__ deg, int E) {
    int i = blockIdx.x * blockDim.x + threadIdx.x;
    if (i < E) atomicAdd(&deg[dst[i]], 1.0f);
}

__global__ void rsqrt_kernel(const float* __restrict__ deg, float* __restrict__ dis, int N) {
    int i = blockIdx.x * blockDim.x + threadIdx.x;
    if (i < N) dis[i] = rsqrtf(deg[i]);   // deg >= 1 always (self-loop)
}

// ---------------------------------------------------------------------------
// Pack W (KxD row-major f32) into per-lane bf16 B-fragments for
// v_wmma_f32_16x16x32_bf16.  B tile = 32(K) x 16(N).  Assumed ISA layout:
//   lane l (n = l&15, half = l>>4): element j (0..7) holds B[half*8 + j][n], j (8..15) holds B[16 + half*8 + j-8][n]
// Fragments stored as [tile][lane][16] contiguous bf16 -> one 32B load/lane.
// ---------------------------------------------------------------------------
__global__ void pack_w_kernel(const float* __restrict__ W, __bf16* __restrict__ Bp, int Dout) {
    int t    = blockIdx.x * blockDim.x + threadIdx.x;
    int lane = t & 31;
    int tile = t >> 5;
    int nTilesN = Dout >> 4;
    int nTiles  = 4 * nTilesN;                 // K = 128 -> 4 k-blocks of 32
    if (tile >= nTiles) return;
    int kb = tile / nTilesN;
    int nt = tile - kb * nTilesN;
    int n     = nt * 16 + (lane & 15);
    int kbase = kb * 32 + (lane >> 4) * 8;
    __bf16* o = Bp + ((size_t)tile * 32 + lane) * 16;
#pragma unroll
    for (int j = 0; j < 16; ++j)
        o[j] = (__bf16)W[(size_t)(kbase + j + (j & 8)) * Dout + n];
}

// ---------------------------------------------------------------------------
// GEMM: C[M x Dout] = A[M x 128] (f32, cast to bf16) x Bp (packed bf16)
// One wave computes a 16 x (NT*16) strip of C: the A fragment is loaded and
// converted ONCE per k-block and reused across NT WMMA ops (NT*8 acc VGPRs).
// 4 waves per block.  K=128 -> 4 k-blocks of 32.
// A-fragment (16-bit A 16x32 ISA layout):
//   lane l: m = mt*16 + (l&15), half = l>>4
//   elements 0..7  -> K = kb*32 + half*8 + 0..7
//   elements 8..15 -> K = kb*32 + 16 + half*8 + 0..7
// ---------------------------------------------------------------------------
template <int NT>
__global__ void __launch_bounds__(128)
gcn_gemm_wmma_bf16(const float* __restrict__ A, const __bf16* __restrict__ Bp,
                   float* __restrict__ C, int Mtiles, int Dout) {
    const int lane = threadIdx.x & 31;
    const int wave = threadIdx.x >> 5;
    const int mt   = blockIdx.x * 4 + wave;
    if (mt >= Mtiles) return;                       // uniform per-wave: EXEC stays all-ones
    const int nt0     = blockIdx.y * NT;
    const int nTilesN = Dout >> 4;
    const int half    = lane >> 4;
    const int l15     = lane & 15;

    const float* arow = A + (size_t)(mt * 16 + l15) * 128;
    v8f acc[NT] = {};

#pragma unroll
    for (int kb = 0; kb < 4; ++kb) {
        const float* ap = arow + kb * 32 + half * 8;
        if (kb < 3) __builtin_prefetch(ap + 32, 0, 3);   // WGP-scope prefetch of next k-block
        const float4 a0 = *(const float4*)(ap + 0);
        const float4 a1 = *(const float4*)(ap + 4);
        const float4 a2 = *(const float4*)(ap + 16);
        const float4 a3 = *(const float4*)(ap + 20);
        v16bf a;
        a[0]  = (__bf16)a0.x; a[1]  = (__bf16)a0.y; a[2]  = (__bf16)a0.z; a[3]  = (__bf16)a0.w;
        a[4]  = (__bf16)a1.x; a[5]  = (__bf16)a1.y; a[6]  = (__bf16)a1.z; a[7]  = (__bf16)a1.w;
        a[8]  = (__bf16)a2.x; a[9]  = (__bf16)a2.y; a[10] = (__bf16)a2.z; a[11] = (__bf16)a2.w;
        a[12] = (__bf16)a3.x; a[13] = (__bf16)a3.y; a[14] = (__bf16)a3.z; a[15] = (__bf16)a3.w;

#pragma unroll
        for (int t = 0; t < NT; ++t) {
            const v16bf b = *(const v16bf*)(
                Bp + (((size_t)kb * nTilesN + (nt0 + t)) * 32 + lane) * 16);
            acc[t] = __builtin_amdgcn_wmma_f32_16x16x32_bf16(
                /*neg_a=*/false, a, /*neg_b=*/false, b,
                /*c_mod=*/(short)0, acc[t], /*reuse_a=*/false, /*reuse_b=*/false);
        }
    }

    // C/D 16x16 f32 layout: VGPR r -> row m = mt*16 + half*8 + r, col n = nt*16 + l15
    float* cbase = C + (size_t)(mt * 16 + half * 8) * Dout + l15;
#pragma unroll
    for (int t = 0; t < NT; ++t) {
        float* crow = cbase + (nt0 + t) * 16;
#pragma unroll
        for (int r = 0; r < 8; ++r)
            crow[(size_t)r * Dout] = acc[t][r];
    }
}

// ---------------------------------------------------------------------------
// Self-loop init: agg[n][:] = h[n][:] * dis[n]^2  (+ optional bias)
// Vectorized float4; d4shift = log2(D/4).
// ---------------------------------------------------------------------------
__global__ void self_init_kernel(const float* __restrict__ h, const float* __restrict__ dis,
                                 const float* __restrict__ bias, float* __restrict__ agg,
                                 int N, int d4shift) {
    int i = blockIdx.x * blockDim.x + threadIdx.x;
    int total = N << d4shift;
    if (i >= total) return;
    int n = i >> d4shift;
    float w = dis[n] * dis[n];
    float4 v = *(const float4*)(h + (size_t)i * 4);
    float4 o;
    o.x = v.x * w; o.y = v.y * w; o.z = v.z * w; o.w = v.w * w;
    if (bias) {
        int col4 = i & ((1 << d4shift) - 1);
        float4 b = *(const float4*)(bias + col4 * 4);
        o.x += b.x; o.y += b.y; o.z += b.z; o.w += b.w;
    }
    *(float4*)(agg + (size_t)i * 4) = o;
}

// ---------------------------------------------------------------------------
// Edge scatter: agg[dst] += h[src] * dis[src]*dis[dst].  1 wave per edge.
// ---------------------------------------------------------------------------
__global__ void scatter128_kernel(const int* __restrict__ src, const int* __restrict__ dst,
                                  const float* __restrict__ dis, const float* __restrict__ h,
                                  float* __restrict__ agg, int E) {
    int e = blockIdx.x * 8 + (threadIdx.x >> 5);
    if (e >= E) return;
    int lane = threadIdx.x & 31;
    int s = src[e], d = dst[e];
    float w = dis[s] * dis[d];
    const float4 v = *(const float4*)(h + (size_t)s * 128 + lane * 4);
    float* o = agg + (size_t)d * 128 + lane * 4;
    atomicAdd(o + 0, v.x * w);
    atomicAdd(o + 1, v.y * w);
    atomicAdd(o + 2, v.z * w);
    atomicAdd(o + 3, v.w * w);
}

__global__ void scatter64_kernel(const int* __restrict__ src, const int* __restrict__ dst,
                                 const float* __restrict__ dis, const float* __restrict__ t,
                                 float* __restrict__ out, int E) {
    int e = blockIdx.x * 8 + (threadIdx.x >> 5);
    if (e >= E) return;
    int lane = threadIdx.x & 31;
    int s = src[e], d = dst[e];
    float w = dis[s] * dis[d];
    const float2 v = *(const float2*)(t + (size_t)s * 64 + lane * 2);
    float* o = out + (size_t)d * 64 + lane * 2;
    atomicAdd(o + 0, v.x * w);
    atomicAdd(o + 1, v.y * w);
}

// ---------------------------------------------------------------------------
// ReLU(agg + b1) -> h2
// ---------------------------------------------------------------------------
__global__ void relu_bias_kernel(const float* __restrict__ agg, const float* __restrict__ bias,
                                 float* __restrict__ h2, int total4, int colmask) {
    int i = blockIdx.x * blockDim.x + threadIdx.x;
    if (i >= total4) return;
    int col4 = i & colmask;
    float4 v = *(const float4*)(agg + (size_t)i * 4);
    float4 b = *(const float4*)(bias + col4 * 4);
    float4 o;
    o.x = fmaxf(v.x + b.x, 0.0f);
    o.y = fmaxf(v.y + b.y, 0.0f);
    o.z = fmaxf(v.z + b.z, 0.0f);
    o.w = fmaxf(v.w + b.w, 0.0f);
    *(float4*)(h2 + (size_t)i * 4) = o;
}

// ---------------------------------------------------------------------------
extern "C" void kernel_launch(void* const* d_in, const int* in_sizes, int n_in,
                              void* d_out, int out_size, void* d_ws, size_t ws_size,
                              hipStream_t stream) {
    const float* x  = (const float*)d_in[0];
    const int*   ei = (const int*)  d_in[1];   // [2, E]
    const float* W1 = (const float*)d_in[2];
    const float* b1 = (const float*)d_in[3];
    const float* W2 = (const float*)d_in[4];
    const float* b2 = (const float*)d_in[5];
    float* out = (float*)d_out;

    const int N = in_sizes[0] / 128;
    const int E = in_sizes[1] / 2;
    const int* src = ei;
    const int* dst = ei + E;

    // Workspace partition (256B aligned)
    char* ws = (char*)d_ws;
    size_t off = 0;
    auto alloc = [&](size_t bytes) -> void* {
        off = (off + 255) & ~(size_t)255;
        void* p = ws + off;
        off += bytes;
        return p;
    };
    float*  deg  = (float*) alloc((size_t)N * 4);
    float*  dis  = (float*) alloc((size_t)N * 4);
    float*  h    = (float*) alloc((size_t)N * 128 * 4);   // h1 output, later reused as h2
    float*  agg1 = (float*) alloc((size_t)N * 128 * 4);   // layer-1 agg, later reused as t
    __bf16* wp1  = (__bf16*)alloc((size_t)128 * 128 * 2);
    __bf16* wp2  = (__bf16*)alloc((size_t)128 * 64 * 2);
    float*  t    = agg1;   // N x 64, fits in agg1 after relu consumes it
    float*  h2   = h;      // overwrite h after scatter1 done

    const int Mtiles = N / 16;                 // 6250
    const int T = 256;

    // 1) degrees + d^-1/2
    deg_init_kernel <<<(N + T - 1) / T, T, 0, stream>>>(deg, N);
    deg_accum_kernel<<<(E + T - 1) / T, T, 0, stream>>>(dst, deg, E);
    rsqrt_kernel    <<<(N + T - 1) / T, T, 0, stream>>>(deg, dis, N);

    // 2) pack weights into bf16 B-fragments
    pack_w_kernel<<<(4 * (128 / 16) * 32 + T - 1) / T, T, 0, stream>>>(W1, wp1, 128);
    pack_w_kernel<<<(4 * (64  / 16) * 32 + T - 1) / T, T, 0, stream>>>(W2, wp2, 64);

    // 3) layer 1: h = x @ W1   (WMMA bf16, f32 accumulate; 4 N-tiles per wave)
    gcn_gemm_wmma_bf16<4><<<dim3((Mtiles + 3) / 4, 2), 128, 0, stream>>>(x, wp1, h, Mtiles, 128);

    // 4) agg1 = self-loop + edge scatter
    self_init_kernel<<<((N * 32) + T - 1) / T, T, 0, stream>>>(h, dis, nullptr, agg1, N, 5);
    scatter128_kernel<<<(E + 7) / 8, T, 0, stream>>>(src, dst, dis, h, agg1, E);

    // 5) h2 = relu(agg1 + b1)
    relu_bias_kernel<<<((N * 32) + T - 1) / T, T, 0, stream>>>(agg1, b1, h2, N * 32, 31);

    // 6) layer 2: t = h2 @ W2
    gcn_gemm_wmma_bf16<4><<<dim3((Mtiles + 3) / 4, 1), 128, 0, stream>>>(h2, wp2, t, Mtiles, 64);

    // 7) out = self-loop + b2, then edge scatter
    self_init_kernel<<<((N * 16) + T - 1) / T, T, 0, stream>>>(t, dis, b2, out, N, 4);
    scatter64_kernel<<<(E + 7) / 8, T, 0, stream>>>(src, dst, dis, t, out, E);
}
